// Attn_83854941487646
// MI455X (gfx1250) — matrix operand order repaired and checked by reading, not compiled, not run on hardware
//
#include <hip/hip_runtime.h>

typedef __attribute__((ext_vector_type(16))) __bf16 v16bf;
typedef __attribute__((ext_vector_type(8)))  __bf16 v8bf;
typedef __attribute__((ext_vector_type(4)))  __bf16 v4bf;
typedef __attribute__((ext_vector_type(8)))  float  v8f;
typedef __attribute__((ext_vector_type(4)))  float  v4f;

#define NB 8
#define NQ 2048
#define NK 2048
#define NH 128
#define KB 32              // keys per tile (== WMMA K for bf16)
#define WAVES 4
#define QB (WAVES * 16)    // 64 query rows per block
#define LOG2E 1.4426950408889634f

__device__ __forceinline__ float fast_exp2(float x) {
#if __has_builtin(__builtin_amdgcn_exp2f)
    return __builtin_amdgcn_exp2f(x);     // v_exp_f32 (hardware 2^x)
#else
    return exp2f(x);
#endif
}
__device__ __forceinline__ float fast_tanh(float x) {
#if __has_builtin(__builtin_amdgcn_tanhf)
    return __builtin_amdgcn_tanhf(x);     // v_tanh_f32 (CDNA5 trans op)
#else
    return tanhf(x);
#endif
}

// Stage one 32x128 history tile into LDS as bf16, row-major [key][h].
// histt already points at the tile (uniform scalar bump per tile), so the
// per-lane part is a loop-invariant 32-bit index -> saddr+voffset loads.
__device__ __forceinline__ void stage_tile(const float* __restrict__ histt,
                                           __bf16* __restrict__ dst_h, int tid) {
    #pragma unroll
    for (int i = 0; i < 8; ++i) {
        int v   = i * 128 + tid;     // float4 index 0..1023
        int key = v >> 5;            // 0..31
        int h4  = (v & 31) * 4;      // h col, lane-contiguous -> coalesced
        v4f f = *(const v4f*)(histt + key * NH + h4);
        v4bf bv;
        #pragma unroll
        for (int j = 0; j < 4; ++j) bv[j] = (__bf16)f[j];
        *(v4bf*)(dst_h + key * NH + h4) = bv;
    }
}

__global__ __launch_bounds__(WAVES * 32)
void Attn_83854941487646_kernel(const float* __restrict__ qsrc,
                                const float* __restrict__ hist,
                                float* __restrict__ outp) {
    __shared__ __align__(32) __bf16 lds_h[2][KB * NH];       // [key][h]  2x8KB

    const int tid    = threadIdx.x;
    const int wave   = tid >> 5;
    const int lane   = tid & 31;
    const int laneLo = lane & 15;
    const int laneHi = lane >> 4;
    const int b      = blockIdx.y;
    const int qbase  = blockIdx.x * QB;

    // ---- Build Q^T B-fragments once: B[k=h][n=query], scaled by log2(e) so
    // softmax exponentials are plain exp2. Lane holds query n=laneLo,
    // elems 0..15 = h in table order (c*32 + 8*laneHi + (j&7) + (j>>3)*16).
    v16bf bQt[4];
    {
        const float* qb = qsrc + ((size_t)b * NQ + qbase + wave * 16) * NH;
        #pragma unroll
        for (int c = 0; c < 4; ++c) {
            #pragma unroll
            for (int j = 0; j < 16; ++j) {
                float q = qb[(size_t)laneLo * NH + c * 32 + laneHi * 8 + (j & 7) + (j >> 3) * 16];
                bQt[c][j] = (__bf16)(fast_tanh(q) * LOG2E);
            }
        }
    }

    // All-ones B fragment: l accumulated as a 9th WMMA accumulator
    // (row-sum(P) = P @ ones, replicated across all 16 output columns).
    v16bf ones;
    #pragma unroll
    for (int j = 0; j < 16; ++j) ones[j] = (__bf16)1.0f;

    v8f acc[8];
    #pragma unroll
    for (int ht = 0; ht < 8; ++ht) acc[ht] = {};
    v8f lacc = {};
    float mrow = -__builtin_inff();   // running max for query = laneLo (log2 domain)

    const float* histb = hist + (size_t)b * NK * NH;
    const int ntiles = NK / KB;

    stage_tile(histb, &lds_h[0][0], tid);
    __syncthreads();

    for (int t = 0; t < ntiles; ++t) {
        const int cur = t & 1;

        // Double-buffered staging: tile t+1's HBM loads overlap t's WMMAs.
        if (t + 1 < ntiles)
            stage_tile(histb + (size_t)(t + 1) * KB * NH, &lds_h[cur ^ 1][0], tid);

        const __bf16* bufh = &lds_h[cur][0];

        // ---- S^T = (H tile) @ (tanh(Q)*log2e)^T : two 16-key sub-tiles.
        // A(16x32) from lds_h rows: contiguous b128 pairs.
        v8f s0 = {};
        v8f s1 = {};
        #pragma unroll
        for (int c = 0; c < 4; ++c) {
            #pragma unroll
            for (int kt = 0; kt < 2; ++kt) {
                const __bf16* ap = bufh + (kt * 16 + laneLo) * NH + c * 32 + laneHi * 8;
                v8bf alo = *(const v8bf*)(ap);
                v8bf ahi = *(const v8bf*)(ap + 16);
                v16bf ah;
                #pragma unroll
                for (int j = 0; j < 8; ++j) { ah[j] = alo[j]; ah[8 + j] = ahi[j]; }
                if (kt == 0)
                    s0 = __builtin_amdgcn_wmma_f32_16x16x32_bf16(false, ah, false, bQt[c],
                                                                 (short)0, s0, false, false);
                else
                    s1 = __builtin_amdgcn_wmma_f32_16x16x32_bf16(false, ah, false, bQt[c],
                                                                 (short)0, s1, false, false);
            }
        }

        // ---- Tile max: each lane holds 16 key-scores of query laneLo.
        float tm = fmaxf(
            fmaxf(fmaxf(fmaxf(s0[0], s0[1]), fmaxf(s0[2], s0[3])),
                  fmaxf(fmaxf(s0[4], s0[5]), fmaxf(s0[6], s0[7]))),
            fmaxf(fmaxf(fmaxf(s1[0], s1[1]), fmaxf(s1[2], s1[3])),
                  fmaxf(fmaxf(s1[4], s1[5]), fmaxf(s1[6], s1[7]))));
        tm = fmaxf(tm, __shfl_xor(tm, 16, 32));

        // ---- Lazy rescale (wave-uniform branch): only when the running max
        // actually grows for some query. Skipped => alpha == 1 for every lane,
        // results bit-identical to always-rescaling.
        if (__any(tm > mrow)) {
            const float mn = fmaxf(mrow, tm);
            const float al = fast_exp2(mrow - mn);
            mrow = mn;
            float alr[8];
            #pragma unroll
            for (int r = 0; r < 8; ++r) alr[r] = __shfl(al, r + 8 * laneHi, 16);
            #pragma unroll
            for (int r = 0; r < 8; ++r) {
                #pragma unroll
                for (int ht = 0; ht < 8; ++ht) acc[ht][r] *= alr[r];
                lacc[r] *= alr[r];
            }
        }

        // ---- P directly in registers: the S^T C-layout element set per lane
        // (query=laneLo, keys {8*laneHi+0..7} u {16+8*laneHi+0..7}) is exactly
        // the 16x32 bf16 A-fragment element set for row=laneLo — no LDS
        // round-trip or cross-lane movement needed.
        v16bf pa;
        #pragma unroll
        for (int r = 0; r < 8; ++r) {
            pa[r]     = (__bf16)fast_exp2(s0[r] - mrow);
            pa[8 + r] = (__bf16)fast_exp2(s1[r] - mrow);
        }

        // ---- l += rowsum(P) via WMMA against ones.
        lacc = __builtin_amdgcn_wmma_f32_16x16x32_bf16(false, pa, false, ones,
                                                       (short)0, lacc, false, false);

        // ---- O += P @ H  (8 h-tiles; B fragment = h column ht*16+laneLo, keys in table order).
        #pragma unroll
        for (int ht = 0; ht < 8; ++ht) {
            v16bf bh;
            #pragma unroll
            for (int j = 0; j < 16; ++j)
                bh[j] = bufh[(laneHi * 8 + (j & 7) + (j >> 3) * 16) * NH + ht * 16 + laneLo];
            acc[ht] = __builtin_amdgcn_wmma_f32_16x16x32_bf16(false, pa, false, bh,
                                                              (short)0, acc[ht], false, false);
        }

        __syncthreads();   // staging of t+1 done; buf[cur] free for t+2
    }

    // ---- Normalize and store (C layout: VGPR r -> row r / r+8, lane -> h col).
    float* ob = outp + ((size_t)b * NQ + qbase + wave * 16) * NH;
    #pragma unroll
    for (int r = 0; r < 8; ++r) {
        const float inv = 1.0f / lacc[r];
        const int row = r + 8 * laneHi;
        #pragma unroll
        for (int ht = 0; ht < 8; ++ht)
            ob[(size_t)row * NH + ht * 16 + laneLo] = acc[ht][r] * inv;
    }
}

extern "C" void kernel_launch(void* const* d_in, const int* in_sizes, int n_in,
                              void* d_out, int out_size, void* d_ws, size_t ws_size,
                              hipStream_t stream) {
    const float* out_state = (const float*)d_in[0];  // [B,Q,H] fp32
    const float* history   = (const float*)d_in[1];  // [B,K,H] fp32
    float* out = (float*)d_out;                      // [B,Q,H] fp32
    dim3 grid(NQ / QB, NB);   // 32 x 8 = 256 workgroups
    dim3 block(WAVES * 32);   // 128 threads = 4 waves (wave32)
    Attn_83854941487646_kernel<<<grid, block, 0, stream>>>(out_state, history, out);
}
